// RAFT_48902497632511
// MI455X (gfx1250) — matrix operand order repaired and checked by reading, not compiled, not run on hardware
//
#include <hip/hip_runtime.h>
#include <hip/hip_bf16.h>
#include <math.h>

// ---------------------------------------------------------------------------
// RAFT forward for MI455X (gfx1250, wave32, WMMA 16x16x32 f16).
// All convs + the all-pairs correlation go through a WMMA implicit-GEMM core:
//   - block tile 64(M) x 64(N) x 32(K), 8 waves (4x2), per-wave 16x32 output
//     (two WMMA sharing one A fragment -> 2x matrix ops per staged LDS byte)
//   - double-buffered LDS staging (f32 -> f16 on load) to overlap VMEM with
//     WMMA; global_prefetch on the weight stream one K-slab ahead
//   - fused bias/activation/scale epilogues
// Pyramid (~100MB) stays L2-resident (192MB) across all 12 GRU iterations.
// ---------------------------------------------------------------------------

typedef __attribute__((ext_vector_type(16))) _Float16 v16h;
typedef __attribute__((ext_vector_type(8)))  _Float16 v8h;
typedef __attribute__((ext_vector_type(8)))  float    v8f;

#define ACT_NONE 0
#define ACT_RELU 1
#define ACT_SIG  2
#define ACT_TANH 3

// problem constants
#define RB   2
#define RH   384
#define RW   512
#define H8   48
#define W8   64
#define HW8  (H8 * W8)        // 3072
#define BHW  (RB * HW8)       // 6144
#define NLVL 4
#define RAD  4
#define NPTS 81               // (2*4+1)^2
#define CCH  (NLVL * NPTS)    // 324
#define HID  128
#define ITERS 12
#define FACT 8

static __device__ __forceinline__ float act_apply(float v, int act) {
    if (act == ACT_RELU) return fmaxf(v, 0.f);
    if (act == ACT_SIG)  return 1.f / (1.f + __expf(-v));
    if (act == ACT_TANH) return tanhf(v);
    return v;
}

// fragment assembly from padded LDS rows (16B-aligned chunk loads)
#define MAKE_AFRAG(AROW, HI, DST)                                              \
    do {                                                                       \
        v8h _lo = *(const v8h*)((AROW) + (HI) * 8);                            \
        v8h _hi = *(const v8h*)((AROW) + 16 + (HI) * 8);                       \
        DST = __builtin_shufflevector(_lo, _hi,                                \
            0,1,2,3,4,5,6,7,8,9,10,11,12,13,14,15);                            \
    } while (0)

#define MAKE_BFRAG(BROW, HI, DST)                                              \
    do {                                                                       \
        v8h _lo = *(const v8h*)((BROW) + (HI) * 8);                            \
        v8h _hi = *(const v8h*)((BROW) + 16 + (HI) * 8);                       \
        DST = __builtin_shufflevector(_lo, _hi,                                \
            0,1,2,3,4,5,6,7,8,9,10,11,12,13,14,15);                            \
    } while (0)

// ---------------------------------------------------------------------------
// Implicit-GEMM conv (NHWC input, HWIO weights), WMMA f16 math, f32 accum.
// ---------------------------------------------------------------------------
__global__ __launch_bounds__(256)
void conv_wmma_kernel(const float* __restrict__ in, const float* __restrict__ wgt,
                      const float* __restrict__ bias, float* __restrict__ out,
                      int Nb, int Hin, int Win, int Cin,
                      int Hout, int Wout, int Cout,
                      int KH, int KW, int stride, int padlo,
                      int act, float scale)
{
    const int M = Nb * Hout * Wout;
    const int K = KH * KW * Cin;

    __shared__ __align__(16) _Float16 As[2][64][40];   // [stage][m][k]
    __shared__ __align__(16) _Float16 Bs[2][64][40];   // [stage][n][k] (transposed)

    const int t    = threadIdx.x;
    const int lane = t & 31;
    const int wv   = t >> 5;
    const int wm   = wv >> 1;        // 0..3  (16-row slab)
    const int wn   = wv & 1;         // 0..1  (32-col slab)
    const int hi   = lane >> 4;      // lane half
    const int m0   = blockIdx.y * 64;
    const int n0   = blockIdx.x * 64;

    // unified loader mapping: row = t>>2 (0..63), 8 contiguous k at (t&3)*8
    const int ra = t >> 2;
    const int ca = (t & 3) * 8;
    const int gm = m0 + ra;          // A row (output pixel)
    const int gn = n0 + ra;          // B row (output channel)
    const bool mvalid = gm < M;
    int b_ = 0, oy = 0, ox = 0;
    if (mvalid) {
        int r = gm;
        ox = r % Wout; r /= Wout;
        oy = r % Hout; b_ = r / Hout;
    }

    v8f acc0 = {0.f, 0.f, 0.f, 0.f, 0.f, 0.f, 0.f, 0.f};
    v8f acc1 = {0.f, 0.f, 0.f, 0.f, 0.f, 0.f, 0.f, 0.f};

#define CONV_LOAD_TILES(K0, S)                                                 \
    do {                                                                       \
        _Pragma("unroll")                                                      \
        for (int i = 0; i < 8; ++i) {                                          \
            const int k = (K0) + ca + i;                                       \
            float v = 0.f;                                                     \
            if (mvalid && k < K) {                                             \
                const int ci = k % Cin;                                        \
                const int kk = k / Cin;                                        \
                const int kw = kk % KW;                                        \
                const int kh = kk / KW;                                        \
                const int iy = oy * stride + kh - padlo;                       \
                const int ix = ox * stride + kw - padlo;                       \
                if (iy >= 0 && iy < Hin && ix >= 0 && ix < Win)                \
                    v = in[(((long)b_ * Hin + iy) * Win + ix) * Cin + ci];     \
            }                                                                  \
            As[S][ra][ca + i] = (_Float16)v;                                   \
        }                                                                      \
        _Pragma("unroll")                                                      \
        for (int i = 0; i < 8; ++i) {                                          \
            const int k = (K0) + ca + i;                                       \
            float v = 0.f;                                                     \
            if (gn < Cout && k < K) v = wgt[(long)k * Cout + gn];              \
            Bs[S][ra][ca + i] = (_Float16)v;                                   \
        }                                                                      \
        if (gn < Cout && (K0) + 32 + ca < K)                                   \
            __builtin_prefetch(wgt + (long)((K0) + 32 + ca) * Cout + gn, 0, 1);\
    } while (0)

#define GEMM_COMPUTE(S)                                                        \
    do {                                                                       \
        const _Float16* arow = &As[S][wm * 16 + (lane & 15)][0];               \
        v16h afrag; MAKE_AFRAG(arow, hi, afrag);                               \
        const _Float16* brow0 = &Bs[S][wn * 32 + (lane & 15)][0];              \
        const _Float16* brow1 = brow0 + 16 * 40;                               \
        v16h bf0; MAKE_BFRAG(brow0, hi, bf0);                                  \
        v16h bf1; MAKE_BFRAG(brow1, hi, bf1);                                  \
        acc0 = __builtin_amdgcn_wmma_f32_16x16x32_f16(                         \
            false, afrag, false, bf0, (short)0, acc0, false, false);           \
        acc1 = __builtin_amdgcn_wmma_f32_16x16x32_f16(                         \
            false, afrag, false, bf1, (short)0, acc1, false, false);           \
    } while (0)

    const int nsteps = (K + 31) / 32;
    CONV_LOAD_TILES(0, 0);
    __syncthreads();
    for (int s = 0; s < nsteps; ++s) {
        const int cur = s & 1;
        if (s + 1 < nsteps) CONV_LOAD_TILES((s + 1) * 32, cur ^ 1);
        GEMM_COMPUTE(cur);
        __syncthreads();
    }
#undef CONV_LOAD_TILES

    // epilogue: D layout M = r + hi*8, N = lane&15 (per 16x16 subtile)
    #pragma unroll
    for (int sub = 0; sub < 2; ++sub) {
        const int nn = n0 + wn * 32 + sub * 16 + (lane & 15);
        if (nn >= Cout) continue;
        const float bv = bias ? bias[nn] : 0.f;
        const v8f& a = sub ? acc1 : acc0;
        #pragma unroll
        for (int r2 = 0; r2 < 8; ++r2) {
            const int mm = m0 + wm * 16 + hi * 8 + r2;
            if (mm < M) {
                float v = (a[r2] + bv) * scale;
                out[(long)mm * Cout + nn] = act_apply(v, act);
            }
        }
    }
}

// ---------------------------------------------------------------------------
// Correlation GEMM (NT): out[b,p,q] = scale * sum_c f1[b,p,c]*f2[b,q,c]
// Same pipelined WMMA core; grid.z = batch.
// ---------------------------------------------------------------------------
__global__ __launch_bounds__(256)
void corr_wmma_kernel(const float* __restrict__ f1, const float* __restrict__ f2,
                      float* __restrict__ out, int M, int N, int K, float scale)
{
    __shared__ __align__(16) _Float16 As[2][64][40];
    __shared__ __align__(16) _Float16 Bs[2][64][40];

    const int z = blockIdx.z;
    const float* A = f1 + (long)z * M * K;
    const float* B = f2 + (long)z * N * K;
    float* O = out + (long)z * M * N;

    const int t    = threadIdx.x;
    const int lane = t & 31;
    const int wv   = t >> 5;
    const int wm   = wv >> 1;
    const int wn   = wv & 1;
    const int hi   = lane >> 4;
    const int m0   = blockIdx.y * 64;
    const int n0   = blockIdx.x * 64;

    const int ra = t >> 2;
    const int ca = (t & 3) * 8;
    const int gm = m0 + ra;
    const int gn = n0 + ra;

    v8f acc0 = {0.f, 0.f, 0.f, 0.f, 0.f, 0.f, 0.f, 0.f};
    v8f acc1 = {0.f, 0.f, 0.f, 0.f, 0.f, 0.f, 0.f, 0.f};

#define CORR_LOAD_TILES(K0, S)                                                 \
    do {                                                                       \
        _Pragma("unroll")                                                      \
        for (int i = 0; i < 8; ++i) {                                          \
            const int k = (K0) + ca + i;                                       \
            As[S][ra][ca + i] =                                                \
                (_Float16)((gm < M && k < K) ? A[(long)gm * K + k] : 0.f);     \
        }                                                                      \
        _Pragma("unroll")                                                      \
        for (int i = 0; i < 8; ++i) {                                          \
            const int k = (K0) + ca + i;                                       \
            Bs[S][ra][ca + i] =                                                \
                (_Float16)((gn < N && k < K) ? B[(long)gn * K + k] : 0.f);     \
        }                                                                      \
        if (gm < M && (K0) + 32 + ca < K)                                      \
            __builtin_prefetch(A + (long)gm * K + (K0) + 32 + ca, 0, 1);       \
    } while (0)

    const int nsteps = (K + 31) / 32;
    CORR_LOAD_TILES(0, 0);
    __syncthreads();
    for (int s = 0; s < nsteps; ++s) {
        const int cur = s & 1;
        if (s + 1 < nsteps) CORR_LOAD_TILES((s + 1) * 32, cur ^ 1);
        GEMM_COMPUTE(cur);
        __syncthreads();
    }
#undef CORR_LOAD_TILES

    #pragma unroll
    for (int sub = 0; sub < 2; ++sub) {
        const int nn = n0 + wn * 32 + sub * 16 + (lane & 15);
        if (nn >= N) continue;
        const v8f& a = sub ? acc1 : acc0;
        #pragma unroll
        for (int r2 = 0; r2 < 8; ++r2) {
            const int mm = m0 + wm * 16 + hi * 8 + r2;
            if (mm < M) O[(long)mm * N + nn] = a[r2] * scale;
        }
    }
}

// ---------------------------------------------------------------------------
// Small support kernels
// ---------------------------------------------------------------------------
__global__ void avgpool2_kernel(const float* __restrict__ in, float* __restrict__ out,
                                int N, int Ho, int Wo)
{
    const int idx = blockIdx.x * blockDim.x + threadIdx.x;
    const long total = (long)N * Ho * Wo;
    if (idx >= total) return;
    const int x = idx % Wo;
    const int y = (idx / Wo) % Ho;
    const int n = idx / (Wo * Ho);
    const int Wi = Wo * 2;
    const float* p = in + ((long)n * (Ho * 2) + 2 * y) * Wi + 2 * x;
    out[idx] = 0.25f * (p[0] + p[1] + p[Wi] + p[Wi + 1]);
}

__global__ void split_ctx_kernel(const float* __restrict__ cout,
                                 float* __restrict__ hidden, float* __restrict__ ctx)
{
    const int idx = blockIdx.x * blockDim.x + threadIdx.x;
    if (idx >= BHW * HID) return;
    const int c = idx % HID;
    const int m = idx / HID;
    hidden[idx] = tanhf(cout[(long)m * 256 + c]);
    ctx[idx]    = fmaxf(cout[(long)m * 256 + HID + c], 0.f);
}

__global__ void init_coords_kernel(float* __restrict__ coords)
{
    const int n = blockIdx.x * blockDim.x + threadIdx.x;
    if (n >= BHW) return;
    coords[n * 2 + 0] = (float)(n % W8);
    coords[n * 2 + 1] = (float)((n / W8) % H8);
}

__global__ void flow_kernel(const float* __restrict__ coords, float* __restrict__ flow)
{
    const int n = blockIdx.x * blockDim.x + threadIdx.x;
    if (n >= BHW) return;
    flow[n * 2 + 0] = coords[n * 2 + 0] - (float)(n % W8);
    flow[n * 2 + 1] = coords[n * 2 + 1] - (float)((n / W8) % H8);
}

__global__ void add_coords_kernel(float* __restrict__ coords, const float* __restrict__ delta)
{
    const int i = blockIdx.x * blockDim.x + threadIdx.x;
    if (i >= BHW * 2) return;
    coords[i] += delta[i];
}

__global__ void concat2_kernel(const float* __restrict__ a, int Ca,
                               const float* __restrict__ b, int Cb,
                               float* __restrict__ out, int M)
{
    const int idx = blockIdx.x * blockDim.x + threadIdx.x;
    const int C = Ca + Cb;
    if (idx >= M * C) return;
    const int c = idx % C;
    const int m = idx / C;
    out[idx] = (c < Ca) ? a[(long)m * Ca + c] : b[(long)m * Cb + (c - Ca)];
}

__global__ void build_rhx_kernel(const float* __restrict__ r, const float* __restrict__ h,
                                 const float* __restrict__ x, float* __restrict__ out)
{
    const int idx = blockIdx.x * blockDim.x + threadIdx.x;
    if (idx >= BHW * 384) return;
    const int c = idx % 384;
    const int m = idx / 384;
    out[idx] = (c < HID) ? r[(long)m * HID + c] * h[(long)m * HID + c]
                         : x[(long)m * 256 + (c - HID)];
}

__global__ void gru_update_kernel(float* __restrict__ h, const float* __restrict__ z,
                                  const float* __restrict__ q)
{
    const int i = blockIdx.x * blockDim.x + threadIdx.x;
    if (i >= BHW * HID) return;
    h[i] = (1.f - z[i]) * h[i] + z[i] * q[i];
}

// bilinear lookup into the 4-level pyramid: cf[n, l*81+p]
__global__ void index_pyramid_kernel(const float* __restrict__ p0, const float* __restrict__ p1,
                                     const float* __restrict__ p2, const float* __restrict__ p3,
                                     const float* __restrict__ coords, float* __restrict__ cf)
{
    const int tid = blockIdx.x * blockDim.x + threadIdx.x;
    if (tid >= BHW * NLVL * NPTS) return;
    const int p = tid % NPTS;
    const int r = tid / NPTS;
    const int l = r & 3;
    const int n = r >> 2;

    const float cx = coords[n * 2 + 0];
    const float cy = coords[n * 2 + 1];
    const float dx = (float)(p % 9 - RAD);
    const float dy = (float)(p / 9 - RAD);
    const float s = 1.f / (float)(1 << l);
    const float x = cx * s + dx;
    const float y = cy * s + dy;

    const int Hl = H8 >> l, Wl = W8 >> l;
    const float* img = ((l == 0) ? p0 : (l == 1) ? p1 : (l == 2) ? p2 : p3)
                       + (long)n * Hl * Wl;

    const float x0 = floorf(x), y0 = floorf(y);
    const float wx = x - x0,   wy = y - y0;

    float v[4];
    #pragma unroll
    for (int ti = 0; ti < 4; ++ti) {
        const float xi = x0 + (float)(ti & 1);
        const float yi = y0 + (float)(ti >> 1);
        float g = 0.f;
        if (xi >= 0.f && xi <= (float)(Wl - 1) && yi >= 0.f && yi <= (float)(Hl - 1))
            g = img[(int)yi * Wl + (int)xi];
        v[ti] = g;
    }
    cf[(long)n * CCH + l * NPTS + p] =
        (1.f - wx) * (1.f - wy) * v[0] + wx * (1.f - wy) * v[1] +
        (1.f - wx) * wy * v[2] + wx * wy * v[3];
}

// convex upsampling: one thread per (b, h, w, fy, fx), both flow channels
__global__ void upsample_kernel(const float* __restrict__ flow, const float* __restrict__ mask,
                                float* __restrict__ out)
{
    const int tid = blockIdx.x * blockDim.x + threadIdx.x;
    if (tid >= RB * H8 * W8 * FACT * FACT) return;
    const int f  = tid & 63;
    const int fx = f & 7, fy = f >> 3;
    int r = tid >> 6;
    const int w = r % W8; r /= W8;
    const int h = r % H8;
    const int b = r / H8;

    const float* mrow = mask + ((long)(b * H8 + h) * W8 + w) * 576 + fy * 8 + fx;
    float mx = -1e30f;
    float e[9];
    #pragma unroll
    for (int ti = 0; ti < 9; ++ti) mx = fmaxf(mx, mrow[ti * 64]);
    float ssum = 0.f;
    #pragma unroll
    for (int ti = 0; ti < 9; ++ti) { e[ti] = __expf(mrow[ti * 64] - mx); ssum += e[ti]; }
    const float inv = 1.f / ssum;

    float a0 = 0.f, a1 = 0.f;
    #pragma unroll
    for (int ti = 0; ti < 9; ++ti) {
        const int hh = h + ti / 3 - 1;
        const int ww = w + ti % 3 - 1;
        if (hh >= 0 && hh < H8 && ww >= 0 && ww < W8) {
            const long fi = ((long)(b * H8 + hh) * W8 + ww) * 2;
            a0 += e[ti] * (float)FACT * flow[fi + 0];
            a1 += e[ti] * (float)FACT * flow[fi + 1];
        }
    }
    const long oi = ((long)(b * RH + h * FACT + fy) * RW + (w * FACT + fx)) * 2;
    out[oi + 0] = a0 * inv;
    out[oi + 1] = a1 * inv;
}

// ---------------------------------------------------------------------------
// Host-side orchestration
// ---------------------------------------------------------------------------
extern "C" void kernel_launch(void* const* d_in, const int* in_sizes, int n_in,
                              void* d_out, int out_size, void* d_ws, size_t ws_size,
                              hipStream_t stream)
{
    (void)in_sizes; (void)n_in; (void)out_size; (void)ws_size;
    const float* image1 = (const float*)d_in[0];
    const float* image2 = (const float*)d_in[1];
    float* outF = (float*)d_out;

    // bump allocator over workspace
    size_t off = 0;
    auto alloc = [&](size_t bytes) -> float* {
        float* p = (float*)((char*)d_ws + off);
        off += (bytes + 255) & ~(size_t)255;
        return p;
    };

    // region0: overlapped scratch (encoder intermediates / per-iteration buffers)
    char* reg0 = (char*)alloc((size_t)100 * 1024 * 1024);
    // phase-1 views
    float* bufE1  = (float*)reg0;                                   // <= 50.4 MB
    float* bufE2  = (float*)(reg0 + (size_t)51 * 1024 * 1024);      // <= 25.2 MB
    float* imgcat = (float*)(reg0 + (size_t)77 * 1024 * 1024);      // 9.44 MB
    // phase-2 views
    size_t o2 = 0;
    auto sub = [&](size_t elems) -> float* {
        float* p = (float*)(reg0 + o2);
        o2 += (elems * 4 + 255) & ~(size_t)255;
        return p;
    };
    float* cf     = sub((size_t)BHW * CCH);
    float* mc1o   = sub((size_t)BHW * 256);
    float* mc2o   = sub((size_t)BHW * 192);
    float* mf1o   = sub((size_t)BHW * 128);
    float* mf2o   = sub((size_t)BHW * 64);
    float* catb   = sub((size_t)BHW * 256);
    float* m_o    = sub((size_t)BHW * 126);
    float* motion = sub((size_t)BHW * 128);
    float* xbuf   = sub((size_t)BHW * 256);
    float* hx     = sub((size_t)BHW * 384);
    float* zb     = sub((size_t)BHW * 128);
    float* rbuf   = sub((size_t)BHW * 128);
    float* rhx    = sub((size_t)BHW * 384);
    float* qb     = sub((size_t)BHW * 128);
    float* fh1o   = sub((size_t)BHW * 256);
    float* mk1o   = sub((size_t)BHW * 256);
    float* upmask = sub((size_t)BHW * 576);

    // persistent buffers
    float* fmaps   = alloc((size_t)2 * BHW * 256 * 4);  // 4 batches of features
    float* coutb   = alloc((size_t)BHW * 256 * 4);
    float* hidden  = alloc((size_t)BHW * HID * 4);
    float* context = alloc((size_t)BHW * HID * 4);
    float* coords1 = alloc((size_t)BHW * 2 * 4);
    float* flowb   = alloc((size_t)BHW * 2 * 4);
    float* deltab  = alloc((size_t)BHW * 2 * 4);
    float* pyr0    = alloc((size_t)BHW * HW8 * 4);              // 75.5 MB
    float* pyr1    = alloc((size_t)BHW * (HW8 / 4) * 4);
    float* pyr2    = alloc((size_t)BHW * (HW8 / 16) * 4);
    float* pyr3    = alloc((size_t)BHW * (HW8 / 64) * 4);

    auto conv = [&](const float* in, int wi, float* out, int Nb, int Hin, int Win,
                    int Cin, int Hout, int Wout, int Cout, int KH, int KW,
                    int stride, int pad, int act, float scale) {
        dim3 g((Cout + 63) / 64, (Nb * Hout * Wout + 63) / 64);
        conv_wmma_kernel<<<g, 256, 0, stream>>>(
            in, (const float*)d_in[wi], (const float*)d_in[wi + 1], out,
            Nb, Hin, Win, Cin, Hout, Wout, Cout, KH, KW, stride, pad, act, scale);
    };
    auto ew = [&](long total) { return dim3((unsigned)((total + 255) / 256)); };

    // ---- feature encoder on concat([image1, image2], axis=0) ----
    const size_t imgBytes = (size_t)RB * RH * RW * 3 * 4;
    hipMemcpyAsync(imgcat, image1, imgBytes, hipMemcpyDeviceToDevice, stream);
    hipMemcpyAsync(imgcat + (size_t)RB * RH * RW * 3, image2, imgBytes,
                   hipMemcpyDeviceToDevice, stream);
    conv(imgcat, 2, bufE1, 4, 384, 512, 3, 192, 256, 64, 3, 3, 2, 0, ACT_RELU, 1.f);
    conv(bufE1,  4, bufE2, 4, 192, 256, 64, 96, 128, 128, 3, 3, 2, 0, ACT_RELU, 1.f);
    conv(bufE2,  6, fmaps, 4, 96, 128, 128, 48, 64, 256, 3, 3, 2, 0, ACT_RELU, 1.f);

    // ---- all-pairs correlation + pyramid ----
    {
        dim3 g(HW8 / 64, HW8 / 64, RB);
        corr_wmma_kernel<<<g, 256, 0, stream>>>(
            fmaps, fmaps + (size_t)RB * HW8 * 256, pyr0, HW8, HW8, 256, 1.f / 16.f);
    }
    avgpool2_kernel<<<ew((long)BHW * 24 * 32), 256, 0, stream>>>(pyr0, pyr1, BHW, 24, 32);
    avgpool2_kernel<<<ew((long)BHW * 12 * 16), 256, 0, stream>>>(pyr1, pyr2, BHW, 12, 16);
    avgpool2_kernel<<<ew((long)BHW * 6 * 8),   256, 0, stream>>>(pyr2, pyr3, BHW, 6, 8);

    // ---- context encoder on image1 ----
    conv(image1, 8, bufE1, 2, 384, 512, 3, 192, 256, 64, 3, 3, 2, 0, ACT_RELU, 1.f);
    conv(bufE1, 10, bufE2, 2, 192, 256, 64, 96, 128, 128, 3, 3, 2, 0, ACT_RELU, 1.f);
    conv(bufE2, 12, coutb, 2, 96, 128, 128, 48, 64, 256, 3, 3, 2, 0, ACT_NONE, 1.f);
    split_ctx_kernel<<<ew((long)BHW * HID), 256, 0, stream>>>(coutb, hidden, context);
    init_coords_kernel<<<ew(BHW), 256, 0, stream>>>(coords1);

    // ---- 12 GRU iterations ----
    for (int it = 0; it < ITERS; ++it) {
        index_pyramid_kernel<<<ew((long)BHW * NLVL * NPTS), 256, 0, stream>>>(
            pyr0, pyr1, pyr2, pyr3, coords1, cf);
        flow_kernel<<<ew(BHW), 256, 0, stream>>>(coords1, flowb);

        // motion encoder
        conv(cf,    14, mc1o, 2, 48, 64, CCH, 48, 64, 256, 1, 1, 1, 0, ACT_RELU, 1.f);
        conv(mc1o,  16, mc2o, 2, 48, 64, 256, 48, 64, 192, 3, 3, 1, 1, ACT_RELU, 1.f);
        conv(flowb, 18, mf1o, 2, 48, 64, 2,   48, 64, 128, 7, 7, 1, 3, ACT_RELU, 1.f);
        conv(mf1o,  20, mf2o, 2, 48, 64, 128, 48, 64, 64,  3, 3, 1, 1, ACT_RELU, 1.f);
        concat2_kernel<<<ew((long)BHW * 256), 256, 0, stream>>>(mc2o, 192, mf2o, 64, catb, BHW);
        conv(catb,  22, m_o,  2, 48, 64, 256, 48, 64, 126, 3, 3, 1, 1, ACT_RELU, 1.f);
        concat2_kernel<<<ew((long)BHW * 128), 256, 0, stream>>>(m_o, 126, flowb, 2, motion, BHW);

        // ConvGRU
        concat2_kernel<<<ew((long)BHW * 256), 256, 0, stream>>>(context, 128, motion, 128, xbuf, BHW);
        concat2_kernel<<<ew((long)BHW * 384), 256, 0, stream>>>(hidden, 128, xbuf, 256, hx, BHW);
        conv(hx, 24, zb,   2, 48, 64, 384, 48, 64, 128, 3, 3, 1, 1, ACT_SIG, 1.f);
        conv(hx, 26, rbuf, 2, 48, 64, 384, 48, 64, 128, 3, 3, 1, 1, ACT_SIG, 1.f);
        build_rhx_kernel<<<ew((long)BHW * 384), 256, 0, stream>>>(rbuf, hidden, xbuf, rhx);
        conv(rhx, 28, qb,  2, 48, 64, 384, 48, 64, 128, 3, 3, 1, 1, ACT_TANH, 1.f);
        gru_update_kernel<<<ew((long)BHW * HID), 256, 0, stream>>>(hidden, zb, qb);

        // flow head
        conv(hidden, 30, fh1o,  2, 48, 64, 128, 48, 64, 256, 3, 3, 1, 1, ACT_RELU, 1.f);
        conv(fh1o,   32, deltab, 2, 48, 64, 256, 48, 64, 2,  3, 3, 1, 1, ACT_NONE, 1.f);
        add_coords_kernel<<<ew((long)BHW * 2), 256, 0, stream>>>(coords1, deltab);

        // upsample mask + convex upsampling
        conv(hidden, 34, mk1o,   2, 48, 64, 128, 48, 64, 256, 3, 3, 1, 1, ACT_RELU, 1.f);
        conv(mk1o,   36, upmask, 2, 48, 64, 256, 48, 64, 576, 1, 1, 1, 0, ACT_NONE, 0.25f);
        flow_kernel<<<ew(BHW), 256, 0, stream>>>(coords1, flowb);
        upsample_kernel<<<ew((long)RB * H8 * W8 * 64), 256, 0, stream>>>(
            flowb, upmask, outF + (size_t)it * RB * RH * RW * 2);
    }
}
